// AlignmentEncoder_39968965656970
// MI455X (gfx1250) — matrix operand order repaired and checked by reading, not compiled, not run on hardware
//
#include <hip/hip_runtime.h>
#include <math.h>

// ---------------------------------------------------------------------------
// Types for CDNA5 WMMA (wave32, 16x16x32 bf16 -> f32)
// ---------------------------------------------------------------------------
typedef __bf16 bf16;
typedef bf16  v16bf __attribute__((ext_vector_type(16)));
typedef float v8f   __attribute__((ext_vector_type(8)));
typedef unsigned int v4u __attribute__((ext_vector_type(4)));

union BV {
    v16bf v;
    bf16  h[16];
    v4u   u[2];
};

union OV {
    bf16 h[8];
    v4u  u;
};

static __device__ __forceinline__ v8f wmma_bf16(v16bf a, v16bf b, v8f c) {
    // 8 args: (neg_a, A, neg_b, B, c_mod, C, reuse_a, reuse_b)
    return __builtin_amdgcn_wmma_f32_16x16x32_bf16(
        false, a, false, b, (short)0, c, false, false);
}

#define CDIV(a, b) (((a) + (b) - 1) / (b))

// Problem constants
#define NB   8
#define MEL  80
#define TXT  512
#define ATT  80
#define TT1  1000
#define TT2  256
#define T1R  1024   // T1 rows padded to mult of 32 (N-pair tiles)
#define CQP  96     // ATT padded to mult of 32

// ---------------------------------------------------------------------------
// Per-batch speaker/emotion bias vectors: kbias[b][c] (c<512), qbias[b][c] (c<80)
// ---------------------------------------------------------------------------
__global__ __launch_bounds__(256)
void bias_kernel(const float* __restrict__ spk, const float* __restrict__ emo,
                 const float* __restrict__ spk_kw, const float* __restrict__ spk_kb,
                 const float* __restrict__ emo_kw, const float* __restrict__ emo_kb,
                 const float* __restrict__ spk_qw, const float* __restrict__ spk_qb,
                 const float* __restrict__ emo_qw, const float* __restrict__ emo_qb,
                 float* __restrict__ kbias, float* __restrict__ qbias)
{
    int idx = blockIdx.x * 256 + threadIdx.x;
    if (idx < NB * TXT) {
        int b = idx >> 9, c = idx & 511;
        const float* sv = spk + b * TXT;
        const float* ev = emo + b * TXT;
        const float* sw = spk_kw + (size_t)c * TXT;
        const float* ew = emo_kw + (size_t)c * TXT;
        float s = spk_kb[c] + emo_kb[c];
        for (int j = 0; j < TXT; j++) s += sv[j] * sw[j] + ev[j] * ew[j];
        kbias[idx] = s;
    } else if (idx < NB * TXT + NB * MEL) {
        int i2 = idx - NB * TXT;
        int b = i2 / MEL, c = i2 % MEL;
        const float* sv = spk + b * TXT;
        const float* ev = emo + b * TXT;
        const float* sw = spk_qw + (size_t)c * TXT;
        const float* ew = emo_qw + (size_t)c * TXT;
        float s = spk_qb[c] + emo_qb[c];
        for (int j = 0; j < TXT; j++) s += sv[j] * sw[j] + ev[j] * ew[j];
        qbias[i2] = s;
    }
}

// f32 weights (O x K) -> bf16 padded (O x Kp), zero-filled K..Kp
__global__ __launch_bounds__(256)
void pack_w_kernel(const float* __restrict__ src, bf16* __restrict__ dst,
                   int K, int Kp, int total)
{
    int idx = blockIdx.x * 256 + threadIdx.x;
    if (idx >= total) return;
    int o = idx / Kp, k = idx - o * Kp;
    float v = (k < K) ? src[(size_t)o * K + k] : 0.0f;
    dst[idx] = (bf16)v;
}

// Transposed im2col for kw=3/pad=1 conv, bias fused, fully initialized:
//   dst[b][t][k] = (t<T && k<Kreal && 0<=t+dt-1<T) ?
//                  bf16(x[b][c][t+dt-1] + biasBC[b][c]) : 0,  k = 3c+dt
__global__ __launch_bounds__(256)
void im2colT3_kernel(const float* __restrict__ x, const float* __restrict__ biasBC,
                     bf16* __restrict__ dst, int C, int T, int Trows,
                     int Kreal, int Kp, int total)
{
    int idx = blockIdx.x * 256 + threadIdx.x;
    if (idx >= total) return;
    int k = idx % Kp;
    int t = (idx / Kp) % Trows;
    int b = idx / (Kp * Trows);
    bf16 v = (bf16)0.0f;
    if (t < T && k < Kreal) {
        int c = k / 3, dt = k - 3 * c;
        int tt = t + dt - 1;
        if (tt >= 0 && tt < T)
            v = (bf16)(x[((size_t)b * C + c) * T + tt] + biasBC[b * C + c]);
    }
    dst[idx] = v;
}

__global__ __launch_bounds__(256)
void zero_kernel(unsigned int* __restrict__ p, int n)
{
    int idx = blockIdx.x * 256 + threadIdx.x;
    if (idx < n) p[idx] = 0u;
}

// k2[b][s] = sum_c kencT[b][s][c]^2  (contiguous bf16 row)
__global__ __launch_bounds__(256)
void k2_kernel(const bf16* __restrict__ kencT, float* __restrict__ k2v)
{
    int idx = blockIdx.x * 256 + threadIdx.x;
    if (idx >= NB * TT2) return;
    const bf16* p = kencT + (size_t)idx * CQP;
    float acc = 0.0f;
    for (int c = 0; c < ATT; c++) { float v = (float)p[c]; acc += v * v; }
    k2v[idx] = acc;
}

// ---------------------------------------------------------------------------
// NT-GEMM with bf16 WMMA: out[b][o][t] = act(sum_k W[o][k] * XT[b][t][k] + bias[o])
// stored TRANSPOSED: outT[b][t][o].  Both operands are K-major -> every
// fragment load is an aligned b128; the C tile stores as one b128 per n-tile.
// Block = 128 threads = 4 waves; wave handles 16(M) x 32(N); no LDS/barriers.
// ---------------------------------------------------------------------------
template<bool RELU>
__global__ __launch_bounds__(128)
void gemm_nt_wmma(const bf16* __restrict__ W,   // O x Kp
                  const bf16* __restrict__ XT,  // per batch: Trows x ldX
                  int ldX, size_t xStride,
                  const float* __restrict__ bias,
                  bf16* __restrict__ outT, int ldO, size_t oStride,
                  int O, int T, int Kp)
{
    const int lane = threadIdx.x & 31;
    const int wv   = threadIdx.x >> 5;
    const int half = lane >> 4;
    const int lm   = lane & 15;
    const int b    = blockIdx.z;
    const int oTile = blockIdx.y * 4 + wv;
    if (oTile * 16 >= O) return;          // wave-uniform, no barriers below
    const int t0 = blockIdx.x * 32;

    // A (16x32, 16-bit): lane<16: row lm, k=[0..15]; lane>=16: k=[16..31]
    const bf16* wrow = W + (size_t)(oTile * 16 + lm) * Kp + half * 16;
    // B (32x16, 16-bit): lane<16: col lm, k=[0..15]; lane>=16: k=[16..31]
    const bf16* xb0 = XT + xStride * b + (size_t)(t0 + lm) * ldX + half * 16;
    const bf16* xb1 = xb0 + (size_t)16 * ldX;

    v8f acc0 = {0.f, 0.f, 0.f, 0.f, 0.f, 0.f, 0.f, 0.f};
    v8f acc1 = {0.f, 0.f, 0.f, 0.f, 0.f, 0.f, 0.f, 0.f};
    #pragma unroll 2
    for (int k0 = 0; k0 < Kp; k0 += 32) {
        BV a, b0, b1;
        a.u[0]  = *reinterpret_cast<const v4u*>(wrow + k0);
        a.u[1]  = *reinterpret_cast<const v4u*>(wrow + k0 + 8);
        b0.u[0] = *reinterpret_cast<const v4u*>(xb0 + k0);
        b0.u[1] = *reinterpret_cast<const v4u*>(xb0 + k0 + 8);
        b1.u[0] = *reinterpret_cast<const v4u*>(xb1 + k0);
        b1.u[1] = *reinterpret_cast<const v4u*>(xb1 + k0 + 8);
        __builtin_prefetch(wrow + k0 + 32, 0, 0);
        acc0 = wmma_bf16(a.v, b0.v, acc0);
        acc1 = wmma_bf16(a.v, b1.v, acc1);
    }

    // bias for this lane's 8 consecutive output channels
    const float* bp = bias + oTile * 16 + half * 8;
    float bv[8];
    #pragma unroll
    for (int i = 0; i < 8; i++) bv[i] = bp[i];

    // C layout: VGPR i -> row m = i + half*8, col n = lm; transposed store is
    // 8 contiguous bf16 -> one b128 per n-tile.
    bf16* obase = outT + oStride * b + oTile * 16 + half * 8;
    #pragma unroll
    for (int p = 0; p < 2; p++) {
        int n = t0 + p * 16 + lm;
        if (n < T) {
            v8f acc = p ? acc1 : acc0;
            OV o;
            #pragma unroll
            for (int i = 0; i < 8; i++) {
                float v = acc[i] + bv[i];
                if (RELU) v = fmaxf(v, 0.0f);
                o.h[i] = (bf16)v;
            }
            *reinterpret_cast<v4u*>(obase + (size_t)n * ldO) = o.u;
        }
    }
}

// ---------------------------------------------------------------------------
// Fused qk GEMM + log_softmax + log-prior + masked softmax.
// Block = 8 waves, handles (b, 16 t-rows x full 256 s).
// qk[t][s] = sum_c qTr[t][c] * kencT[s][c]  (K = 96 padded, all-b128 loads)
// logits l[s] = 2*TEMP*qk - TEMP*k2[s]      (row-constant -TEMP*q2 cancels)
// ---------------------------------------------------------------------------
__global__ __launch_bounds__(256)
void qk_softmax_kernel(const bf16* __restrict__ qT, const bf16* __restrict__ kT,
                       const float* __restrict__ k2v, const float* __restrict__ prior,
                       const unsigned char* __restrict__ mask,
                       float* __restrict__ attnOut, float* __restrict__ logpOut)
{
    __shared__ float lg[16][TT2];    // 16 KB
    const int tid  = threadIdx.x;
    const int wv   = tid >> 5;
    const int lane = tid & 31;
    const int half = lane >> 4;
    const int lm   = lane & 15;
    const int b     = blockIdx.y;
    const int tBase = blockIdx.x * 16;

    const bf16* arow  = qT + (size_t)b * (T1R * CQP) + (size_t)(tBase + lm) * CQP + half * 16;
    const int s0 = wv * 32 + lm;     // n-tile 2*wv
    const int s1 = s0 + 16;          // n-tile 2*wv+1
    const bf16* brow0 = kT + (size_t)b * (TT2 * CQP) + (size_t)s0 * CQP + half * 16;
    const bf16* brow1 = brow0 + (size_t)16 * CQP;

    v8f acc0 = {0.f, 0.f, 0.f, 0.f, 0.f, 0.f, 0.f, 0.f};
    v8f acc1 = {0.f, 0.f, 0.f, 0.f, 0.f, 0.f, 0.f, 0.f};
    #pragma unroll
    for (int k0 = 0; k0 < CQP; k0 += 32) {
        BV a, b0, b1;
        a.u[0]  = *reinterpret_cast<const v4u*>(arow + k0);
        a.u[1]  = *reinterpret_cast<const v4u*>(arow + k0 + 8);
        b0.u[0] = *reinterpret_cast<const v4u*>(brow0 + k0);
        b0.u[1] = *reinterpret_cast<const v4u*>(brow0 + k0 + 8);
        b1.u[0] = *reinterpret_cast<const v4u*>(brow1 + k0);
        b1.u[1] = *reinterpret_cast<const v4u*>(brow1 + k0 + 8);
        acc0 = wmma_bf16(a.v, b0.v, acc0);
        acc1 = wmma_bf16(a.v, b1.v, acc1);
    }
    #pragma unroll
    for (int i = 0; i < 8; i++) {
        lg[i + half * 8][s0] = acc0[i];
        lg[i + half * 8][s1] = acc1[i];
    }
    __syncthreads();

    const float TEMP = 0.0005f;
    const float NEG_INF = -__builtin_inff();
    #pragma unroll
    for (int rr = 0; rr < 2; rr++) {
        int r = wv * 2 + rr;     // wave-uniform
        int t = tBase + r;
        if (t >= TT1) continue;
        float v[8];
        float mx = -3.4e38f;
        #pragma unroll
        for (int it = 0; it < 8; it++) {
            int s = lane + it * 32;
            float val = 2.0f * TEMP * lg[r][s] - TEMP * k2v[b * TT2 + s];
            v[it] = val;
            mx = fmaxf(mx, val);
        }
        #pragma unroll
        for (int off = 16; off > 0; off >>= 1) mx = fmaxf(mx, __shfl_xor(mx, off, 32));
        float sum = 0.0f;
        #pragma unroll
        for (int it = 0; it < 8; it++) sum += expf(v[it] - mx);
        #pragma unroll
        for (int off = 16; off > 0; off >>= 1) sum += __shfl_xor(sum, off, 32);
        float lse = mx + logf(sum);

        size_t rowOff = ((size_t)b * TT1 + t) * TT2;
        float mval[8];
        float mx2 = NEG_INF;
        #pragma unroll
        for (int it = 0; it < 8; it++) {
            int s = lane + it * 32;
            float l = v[it] - lse + logf(prior[rowOff + s] + 1e-8f);
            logpOut[rowOff + s] = l;
            float mm = mask[b * TT2 + s] ? NEG_INF : l;
            mval[it] = mm;
            mx2 = fmaxf(mx2, mm);
        }
        #pragma unroll
        for (int off = 16; off > 0; off >>= 1) mx2 = fmaxf(mx2, __shfl_xor(mx2, off, 32));
        float s2 = 0.0f;
        #pragma unroll
        for (int it = 0; it < 8; it++) s2 += expf(mval[it] - mx2);
        #pragma unroll
        for (int off = 16; off > 0; off >>= 1) s2 += __shfl_xor(s2, off, 32);
        float inv = 1.0f / s2;
        #pragma unroll
        for (int it = 0; it < 8; it++) {
            int s = lane + it * 32;
            attnOut[rowOff + s] = expf(mval[it] - mx2) * inv;
        }
    }
}

// ---------------------------------------------------------------------------
// Host launcher
// ---------------------------------------------------------------------------
extern "C" void kernel_launch(void* const* d_in, const int* in_sizes, int n_in,
                              void* d_out, int out_size, void* d_ws, size_t ws_size,
                              hipStream_t stream)
{
    (void)in_sizes; (void)n_in; (void)out_size; (void)ws_size;
    const float* queries = (const float*)d_in[0];
    const float* keys    = (const float*)d_in[1];
    const float* prior   = (const float*)d_in[2];
    const float* spk     = (const float*)d_in[3];
    const float* emo     = (const float*)d_in[4];
    const float* kw1 = (const float*)d_in[5];
    const float* kb1 = (const float*)d_in[6];
    const float* kw2 = (const float*)d_in[7];
    const float* kb2 = (const float*)d_in[8];
    const float* qw1 = (const float*)d_in[9];
    const float* qb1 = (const float*)d_in[10];
    const float* qw2 = (const float*)d_in[11];
    const float* qb2 = (const float*)d_in[12];
    const float* qw3 = (const float*)d_in[13];
    const float* qb3 = (const float*)d_in[14];
    const float* spk_kw = (const float*)d_in[15];
    const float* spk_kb = (const float*)d_in[16];
    const float* spk_qw = (const float*)d_in[17];
    const float* spk_qb = (const float*)d_in[18];
    const float* emo_kw = (const float*)d_in[19];
    const float* emo_kb = (const float*)d_in[20];
    const float* emo_qw = (const float*)d_in[21];
    const float* emo_qb = (const float*)d_in[22];
    const unsigned char* mask = (const unsigned char*)d_in[23];   // jnp bool, 1 byte

    float* attnOut = (float*)d_out;
    float* logpOut = attnOut + (size_t)NB * TT1 * TT2;

    // --- workspace carve (256-B aligned) ---
    char* base = (char*)d_ws;
    size_t off = 0;
    auto alloc = [&](size_t bytes) -> void* {
        off = (off + 255) & ~(size_t)255;
        void* p = base + off;
        off += bytes;
        return p;
    };
    float* kbias = (float*)alloc((size_t)NB * TXT * 4);
    float* qbias = (float*)alloc((size_t)NB * MEL * 4);
    bf16* kw1p = (bf16*)alloc((size_t)1024 * 1536 * 2);
    bf16* kw2p = (bf16*)alloc((size_t)80 * 1024 * 2);
    bf16* qw1p = (bf16*)alloc((size_t)160 * 256 * 2);
    bf16* qw2p = (bf16*)alloc((size_t)80 * 160 * 2);
    bf16* qw3p = (bf16*)alloc((size_t)80 * CQP * 2);
    bf16* ic_k1  = (bf16*)alloc((size_t)NB * TT2 * 1536 * 2);   // im2colT keys
    bf16* ic_q1  = (bf16*)alloc((size_t)NB * T1R * 256 * 2);    // im2colT queries
    bf16* kmidT  = (bf16*)alloc((size_t)NB * TT2 * 1024 * 2);
    bf16* kencT  = (bf16*)alloc((size_t)NB * TT2 * CQP * 2);
    bf16* qmid1T = (bf16*)alloc((size_t)NB * T1R * 160 * 2);
    bf16* qmid2T = (bf16*)alloc((size_t)NB * T1R * CQP * 2);
    bf16* qTr    = (bf16*)alloc((size_t)NB * T1R * CQP * 2);
    float* k2v   = (float*)alloc((size_t)NB * TT2 * 4);

    // 1. speaker/emotion per-batch bias vectors
    bias_kernel<<<CDIV(NB*TXT + NB*MEL, 256), 256, 0, stream>>>(
        spk, emo, spk_kw, spk_kb, emo_kw, emo_kb, spk_qw, spk_qb, emo_qw, emo_qb,
        kbias, qbias);

    // 2. pack weights to bf16 (K padded to mult of 32, zero-filled)
    pack_w_kernel<<<CDIV(1024*1536, 256), 256, 0, stream>>>(kw1, kw1p, 1536, 1536, 1024*1536);
    pack_w_kernel<<<CDIV(80*1024, 256), 256, 0, stream>>>(kw2, kw2p, 1024, 1024, 80*1024);
    pack_w_kernel<<<CDIV(160*256, 256), 256, 0, stream>>>(qw1, qw1p, 240, 256, 160*256);
    pack_w_kernel<<<CDIV(80*160, 256), 256, 0, stream>>>(qw2, qw2p, 160, 160, 80*160);
    pack_w_kernel<<<CDIV(80*CQP, 256), 256, 0, stream>>>(qw3, qw3p, 80, CQP, 80*CQP);

    // 3. transposed, bias-fused im2col panels (fully initialized incl. pads)
    im2colT3_kernel<<<CDIV(NB*TT2*1536, 256), 256, 0, stream>>>(
        keys, kbias, ic_k1, TXT, TT2, TT2, 1536, 1536, NB*TT2*1536);
    im2colT3_kernel<<<CDIV(NB*T1R*256, 256), 256, 0, stream>>>(
        queries, qbias, ic_q1, MEL, TT1, T1R, 240, 256, NB*T1R*256);

    // 4. zero-fill buffers whose K-padding is consumed as real GEMM entries
    {
        int w1 = (int)((size_t)NB * TT2 * CQP * 2 / 4);   // kencT
        int w2 = (int)((size_t)NB * T1R * CQP * 2 / 4);   // qmid2T / qTr
        zero_kernel<<<CDIV(w1, 256), 256, 0, stream>>>((unsigned int*)kencT, w1);
        zero_kernel<<<CDIV(w2, 256), 256, 0, stream>>>((unsigned int*)qmid2T, w2);
        zero_kernel<<<CDIV(w2, 256), 256, 0, stream>>>((unsigned int*)qTr, w2);
    }

    // 5. keys path: conv(512->1024, kw3)+relu ; conv(1024->80, 1x1)
    gemm_nt_wmma<true ><<<dim3(TT2/32, 16, NB), dim3(128), 0, stream>>>(
        kw1p, ic_k1, 1536, (size_t)TT2*1536, kb1,
        kmidT, 1024, (size_t)TT2*1024, 1024, TT2, 1536);
    gemm_nt_wmma<false><<<dim3(TT2/32, 2, NB), dim3(128), 0, stream>>>(
        kw2p, kmidT, 1024, (size_t)TT2*1024, kb2,
        kencT, CQP, (size_t)TT2*CQP, ATT, TT2, 1024);

    // 6. queries path: conv(80->160,kw3)+relu ; conv(160->80)+relu ; conv(80->80)
    gemm_nt_wmma<true ><<<dim3(T1R/32, 3, NB), dim3(128), 0, stream>>>(
        qw1p, ic_q1, 256, (size_t)T1R*256, qb1,
        qmid1T, 160, (size_t)T1R*160, 160, TT1, 256);
    gemm_nt_wmma<true ><<<dim3(T1R/32, 2, NB), dim3(128), 0, stream>>>(
        qw2p, qmid1T, 160, (size_t)T1R*160, qb2,
        qmid2T, CQP, (size_t)T1R*CQP, ATT, TT1, 160);
    gemm_nt_wmma<false><<<dim3(T1R/32, 2, NB), dim3(128), 0, stream>>>(
        qw3p, qmid2T, CQP, (size_t)T1R*CQP, qb3,
        qTr, CQP, (size_t)T1R*CQP, ATT, TT1, CQP);

    // 7. k2[b][s] from transposed kenc (same values qk consumes)
    k2_kernel<<<CDIV(NB*TT2, 256), 256, 0, stream>>>(kencT, k2v);

    // 8. fused qk GEMM + log_softmax + log-prior + masked softmax
    qk_softmax_kernel<<<dim3(CDIV(TT1,16), NB), dim3(256), 0, stream>>>(
        qTr, kencT, k2v, prior, mask, attnOut, logpOut);
}
